// SelfAttention_38929583571421
// MI455X (gfx1250) — compile-verified, matrix operand order checked by reading, not run on hardware
//
#include <hip/hip_runtime.h>
#include <hip/hip_bf16.h>
#include <cmath>

typedef __attribute__((ext_vector_type(4)))  __bf16 v4bf;
typedef __attribute__((ext_vector_type(8)))  __bf16 v8bf;
typedef __attribute__((ext_vector_type(16))) __bf16 v16bf;
typedef __attribute__((ext_vector_type(8)))  float  v8f;

// ---- CDNA5 async copy: global -> LDS, 16 bytes per lane, ASYNCcnt-tracked ----
static __device__ inline void async_b128(unsigned lds_off, const void* g) {
  asm volatile("global_load_async_to_lds_b128 %0, %1, off"
               :: "v"(lds_off), "v"((unsigned long long)(uintptr_t)g)
               : "memory");
}
static __device__ inline void wait_async0() {
  asm volatile("s_wait_asynccnt 0x0" ::: "memory");
}
static __device__ inline unsigned lds_off(const void* p) {
  return (unsigned)(uintptr_t)p;  // low 32 bits of generic ptr == LDS byte offset
}

// Load one WMMA operand fragment for 16x16x32 bf16: per-lane data is two
// contiguous 8-element (16B) runs at +0 and +16 elements from the lane base.
static __device__ inline v16bf load_frag(const __bf16* p) {
  v8bf lo = *(const v8bf*)(p);
  v8bf hi = *(const v8bf*)(p + 16);
  return __builtin_shufflevector(lo, hi, 0,1,2,3,4,5,6,7,8,9,10,11,12,13,14,15);
}

static __device__ inline v8f wmma_bf16(v16bf a, v16bf b, v8f c) {
  return __builtin_amdgcn_wmma_f32_16x16x32_bf16(
      false, a, false, b, (short)0, c, false, false);
}

// ---------------------------------------------------------------------------
// Pre-pass: fp32 -> bf16 elementwise (x) and fp32 -> bf16 transpose (weights)
// ---------------------------------------------------------------------------
__global__ __launch_bounds__(256) void cvt_bf16(const float* __restrict__ in,
                                                __bf16* __restrict__ out) {
  const int i = blockIdx.x * 256 + threadIdx.x;
  const float4 f = ((const float4*)in)[i];
  v4bf p = { (__bf16)f.x, (__bf16)f.y, (__bf16)f.z, (__bf16)f.w };
  ((v4bf*)out)[i] = p;
}

// Wt[n][k] = (bf16) W[k][n], 1024x1024
__global__ __launch_bounds__(256) void cvt_wT(const float* __restrict__ W,
                                              __bf16* __restrict__ Wt) {
  const int idx = blockIdx.x * 256 + threadIdx.x;
  const int n = idx >> 8, k4 = (idx & 255) * 4;
  v4bf p = { (__bf16)W[(size_t)(k4 + 0) * 1024 + n],
             (__bf16)W[(size_t)(k4 + 1) * 1024 + n],
             (__bf16)W[(size_t)(k4 + 2) * 1024 + n],
             (__bf16)W[(size_t)(k4 + 3) * 1024 + n] };
  *(v4bf*)&Wt[(size_t)n * 1024 + k4] = p;
}

// ---------------------------------------------------------------------------
// GEMM: C[8192,1024] = A[8192,1024](bf16) @ Wt^T + bias, Wt is [N][K] bf16.
// Block tile 128x128, 8 waves (2Mx4N), each wave 64x32 via 4x2 WMMA accs.
// Double-buffered LDS filled by async copies (ASYNCcnt), no ds_stores.
//   STORE_MODE: 0 = bf16 row-major, 1 = bf16 transposed [1024][8192], 2 = f32
// ---------------------------------------------------------------------------
template<int STORE_MODE>
__global__ __launch_bounds__(256) void gemm128(const __bf16* __restrict__ A,
                                               const __bf16* __restrict__ Wt,
                                               const float* __restrict__ bias,
                                               void* __restrict__ Outp) {
  constexpr int K = 1024, N = 1024;
  __shared__ __bf16 ldsA[2][128][40];   // [m][k], padded rows stay 16B-aligned
  __shared__ __bf16 ldsB[2][128][40];   // [n][k]
  const int tid = threadIdx.x;
  const int lane = tid & 31, w = tid >> 5;
  const int lane15 = lane & 15, hi8 = (lane >> 4) * 8;
  const int wm = w >> 2, wn = w & 3;
  const size_t r0 = (size_t)blockIdx.y * 128;
  const int n0 = blockIdx.x * 128;

  v8f acc[4][2] = {};

  auto stage = [&](int buf, int kc) {
    // 128 rows x 32 cols bf16 per tile = 512 16B-chunks; 2 per thread per tile
    for (int i = tid; i < 512; i += 256) {
      const int row = i >> 2, c8 = (i & 3) * 8;
      async_b128(lds_off(&ldsA[buf][row][c8]), &A[(r0 + row) * K + kc + c8]);
      async_b128(lds_off(&ldsB[buf][row][c8]),
                 &Wt[(size_t)(n0 + row) * K + kc + c8]);
    }
  };

  stage(0, 0);
  int buf = 0;
  for (int kc = 0; kc < K; kc += 32) {
    wait_async0();        // my async fills of `buf` have landed
    __syncthreads();      // everyone's have landed
    if (kc + 32 < K) stage(buf ^ 1, kc + 32);  // overlap next fill with math

    v16bf af[4], bf[2];
    #pragma unroll
    for (int i = 0; i < 4; ++i)
      af[i] = load_frag(&ldsA[buf][wm * 64 + i * 16 + lane15][hi8]);
    #pragma unroll
    for (int j = 0; j < 2; ++j)
      bf[j] = load_frag(&ldsB[buf][wn * 32 + j * 16 + lane15][hi8]);
    #pragma unroll
    for (int i = 0; i < 4; ++i)
      #pragma unroll
      for (int j = 0; j < 2; ++j)
        acc[i][j] = wmma_bf16(af[i], bf[j], acc[i][j]);
    buf ^= 1;
  }

  // ---- epilogue: bias + store ----
  #pragma unroll
  for (int i = 0; i < 4; ++i) {
    #pragma unroll
    for (int j = 0; j < 2; ++j) {
      const int col = n0 + wn * 32 + j * 16 + lane15;
      const float bv = bias[col];
      const size_t rowbase = r0 + wm * 64 + i * 16 + hi8;  // 8 consecutive rows
      if constexpr (STORE_MODE == 0) {
        __bf16* Out = (__bf16*)Outp;
        #pragma unroll
        for (int r = 0; r < 8; ++r)
          Out[(rowbase + r) * N + col] = (__bf16)(acc[i][j][r] + bv);
      } else if constexpr (STORE_MODE == 1) {
        __bf16* Out = (__bf16*)Outp;  // [N][8192]; 8 rows contiguous -> b128
        v8bf pk;
        #pragma unroll
        for (int r = 0; r < 8; ++r) pk[r] = (__bf16)(acc[i][j][r] + bv);
        *(v8bf*)&Out[(size_t)col * 8192 + rowbase] = pk;
      } else {
        float* Out = (float*)Outp;
        #pragma unroll
        for (int r = 0; r < 8; ++r)
          Out[(rowbase + r) * N + col] = acc[i][j][r] + bv;
      }
    }
  }
}

// ---------------------------------------------------------------------------
// Flash attention: one workgroup per (batch, 16 query rows).
// 8 waves: wave w computes the score tile vs keys [kb+16w, +16), then owns
// output cols [128w, +128). K fragments straight from global (L2-resident);
// V read from pre-transposed Vt[1024][8192] so B-frags are contiguous b128.
// ---------------------------------------------------------------------------
__global__ __launch_bounds__(256) void attn_kernel(const __bf16* __restrict__ Q,
                                                   const __bf16* __restrict__ Km,
                                                   const __bf16* __restrict__ Vt,
                                                   __bf16* __restrict__ Ctx) {
  constexpr int D = 1024, S = 2048;
  __shared__ __bf16 ldsQ[16][1032];
  __shared__ float  ldsS[16][128];
  __shared__ __bf16 ldsP[16][136];
  __shared__ float rowM[16], rowL[16], rowC[16];

  const int tid = threadIdx.x;
  const int lane = tid & 31, w = tid >> 5;
  const int lane15 = lane & 15, hi8 = (lane >> 4) * 8;
  const int batch = blockIdx.y;
  const size_t bs0 = (size_t)batch * S + (size_t)blockIdx.x * 16;

  // async-stage Q block (16 x 1024 bf16 = 32KB), reused by all waves
  for (int i = tid; i < 16 * (D / 8); i += 256) {
    const int rr = i >> 7, c8 = (i & 127) * 8;
    async_b128(lds_off(&ldsQ[rr][c8]), &Q[(bs0 + rr) * D + c8]);
  }
  if (tid < 16) { rowM[tid] = -INFINITY; rowL[tid] = 0.f; }
  wait_async0();
  __syncthreads();

  v8f o[8] = {};                 // 16 q-rows x 128 cols per wave
  const int c0 = w * 128;
  const size_t kvbase = (size_t)batch * S;

  for (int kb = 0; kb < S; kb += 128) {
    // ---- scores: 16x16 tile = Qblk @ K[kb+16w..]^T, reduce over D=1024 ----
    v8f s = {};
    const __bf16* Kp = Km + (kvbase + kb + w * 16 + lane15) * D + hi8;
    for (int kc = 0; kc < D; kc += 32) {
      v16bf aq = load_frag(&ldsQ[lane15][kc + hi8]);
      v16bf bk = load_frag(Kp + kc);
      s = wmma_bf16(aq, bk, s);
    }
    #pragma unroll
    for (int r = 0; r < 8; ++r)
      ldsS[r + hi8][w * 16 + lane15] = s[r] * 0.03125f;  // 1/sqrt(1024)
    __syncthreads();

    // ---- online softmax over this 128-key block (fp32) ----
    if (tid < 16) {
      const int row = tid;
      float rmax = -INFINITY;
      for (int c = 0; c < 128; ++c) rmax = fmaxf(rmax, ldsS[row][c]);
      const float mnew = fmaxf(rowM[row], rmax);
      const float corr = __expf(rowM[row] - mnew);
      float sum = 0.f;
      for (int c = 0; c < 128; ++c) {
        const float p = __expf(ldsS[row][c] - mnew);
        ldsP[row][c] = (__bf16)p;
        sum += p;
      }
      rowL[row] = rowL[row] * corr + sum;
      rowM[row] = mnew;
      rowC[row] = corr;
    }
    __syncthreads();

    // ---- rescale O, then O += P @ V ----
    float cf[8];
    #pragma unroll
    for (int r = 0; r < 8; ++r) cf[r] = rowC[r + hi8];
    #pragma unroll
    for (int t = 0; t < 8; ++t)
      #pragma unroll
      for (int r = 0; r < 8; ++r) o[t][r] *= cf[r];

    for (int ks = 0; ks < 128; ks += 32) {
      v16bf ap = load_frag(&ldsP[lane15][ks + hi8]);
      const __bf16* Vp = Vt + kvbase + kb + ks + hi8;  // + col*8192 below
      #pragma unroll
      for (int t = 0; t < 8; ++t) {
        v16bf bv = load_frag(Vp + (size_t)(c0 + t * 16 + lane15) * 8192);
        o[t] = wmma_bf16(ap, bv, o[t]);
      }
    }
  }

  // ---- finalize: divide by l, store ctx as bf16 row-major ----
  float inv[8];
  #pragma unroll
  for (int r = 0; r < 8; ++r) inv[r] = 1.f / rowL[r + hi8];
  #pragma unroll
  for (int t = 0; t < 8; ++t) {
    const int col = c0 + t * 16 + lane15;
    #pragma unroll
    for (int r = 0; r < 8; ++r)
      Ctx[(bs0 + r + hi8) * D + col] = (__bf16)(o[t][r] * inv[r]);
  }
}

extern "C" void kernel_launch(void* const* d_in, const int* in_sizes, int n_in,
                              void* d_out, int out_size, void* d_ws, size_t ws_size,
                              hipStream_t stream) {
  (void)in_sizes; (void)n_in; (void)out_size; (void)ws_size;
  const float* x  = (const float*)d_in[0];
  const float* Wq = (const float*)d_in[1];
  const float* bq = (const float*)d_in[2];
  const float* Wk = (const float*)d_in[3];
  const float* bk = (const float*)d_in[4];
  const float* Wv = (const float*)d_in[5];
  const float* bv = (const float*)d_in[6];
  const float* Wo = (const float*)d_in[7];
  const float* bo = (const float*)d_in[8];
  float* out = (float*)d_out;

  char* ws = (char*)d_ws;
  const size_t MATB = (size_t)8192 * 1024 * sizeof(__bf16);  // 16 MB
  const size_t WTB  = (size_t)1024 * 1024 * sizeof(__bf16);  //  2 MB
  __bf16* xb  = (__bf16*)(ws);
  __bf16* WqT = (__bf16*)(ws + MATB);
  __bf16* WkT = (__bf16*)(ws + MATB + 1 * WTB);
  __bf16* WvT = (__bf16*)(ws + MATB + 2 * WTB);
  __bf16* WoT = (__bf16*)(ws + MATB + 3 * WTB);
  __bf16* Qb  = (__bf16*)(ws + MATB + 4 * WTB);
  __bf16* Kb  = (__bf16*)(ws + 2 * MATB + 4 * WTB);
  __bf16* Vtb = (__bf16*)(ws + 3 * MATB + 4 * WTB);  // [1024][8192]
  __bf16* Cx  = (__bf16*)(ws + 4 * MATB + 4 * WTB);

  const dim3 b(256);
  hipLaunchKernelGGL(cvt_bf16, dim3(8192), b, 0, stream, x, xb);
  hipLaunchKernelGGL(cvt_wT, dim3(1024), b, 0, stream, Wq, WqT);
  hipLaunchKernelGGL(cvt_wT, dim3(1024), b, 0, stream, Wk, WkT);
  hipLaunchKernelGGL(cvt_wT, dim3(1024), b, 0, stream, Wv, WvT);
  hipLaunchKernelGGL(cvt_wT, dim3(1024), b, 0, stream, Wo, WoT);

  const dim3 g(8, 64);
  hipLaunchKernelGGL((gemm128<0>), g, b, 0, stream, xb, WqT, bq, (void*)Qb);
  hipLaunchKernelGGL((gemm128<0>), g, b, 0, stream, xb, WkT, bk, (void*)Kb);
  hipLaunchKernelGGL((gemm128<1>), g, b, 0, stream, xb, WvT, bv, (void*)Vtb);
  hipLaunchKernelGGL(attn_kernel, dim3(128, 4), b, 0, stream, Qb, Kb, Vtb, Cx);
  hipLaunchKernelGGL((gemm128<2>), g, b, 0, stream, Cx, WoT, bo, (void*)out);
}
